// LSA_2740189135146
// MI455X (gfx1250) — matrix operand order repaired and checked by reading, not compiled, not run on hardware
//
#include <hip/hip_runtime.h>
#include <hip/hip_bf16.h>
#include <math.h>

// ---------------------------------------------------------------------------
// MI455X (gfx1250) fused SigmaReparam attention, bf16 WMMA path.
// wave32; v_wmma_f32_16x16x32_bf16; flash-attention so S[2048,2048] never
// hits memory. All scratch (~52 MB) fits in the 192 MB L2.
// ---------------------------------------------------------------------------

typedef __attribute__((ext_vector_type(16))) __bf16 v16bf;
typedef __attribute__((ext_vector_type(8)))  __bf16 v8bf;
typedef __attribute__((ext_vector_type(8)))  float  v8f;
typedef __bf16 bf16;

#define B_  4
#define N_  2048
#define DIM 512
#define HEADS 8
#define DH  64
#define INNER 512
#define ROWS (B_*N_)          // 8192
#define NEG_MAX (-3.4028234663852886e38f)

__device__ __forceinline__ bf16 f2bf(float f) {
  unsigned u = __builtin_bit_cast(unsigned, f);
  unsigned r = u + 0x7FFFu + ((u >> 16) & 1u);   // round-to-nearest-even
  unsigned short h = (unsigned short)(r >> 16);
  return __builtin_bit_cast(__bf16, h);
}

__device__ __forceinline__ v16bf make_a(v8bf lo, v8bf hi) {
  v16bf r;
#pragma unroll
  for (int i = 0; i < 8; i++) { r[i] = lo[i]; r[i + 8] = hi[i]; }
  return r;
}

__device__ __forceinline__ v8f wmma_bf16(v16bf a, v16bf b, v8f c) {
  return __builtin_amdgcn_wmma_f32_16x16x32_bf16(false, a, false, b,
                                                 (short)0, c, false, false);
}

// ---------------------------------------------------------------------------
// K0: spectral-norm scale = sigma / (v . W u2), one power-iteration step.
// Single workgroup; tiny (<= 1536x512).
// ---------------------------------------------------------------------------
__global__ __launch_bounds__(256)
void sn_scale_kernel(const float* __restrict__ W, const float* __restrict__ u,
                     const float* __restrict__ sigma, float* __restrict__ out,
                     int rows, int cols) {
  __shared__ float sv[1536];
  __shared__ float su[512];
  __shared__ float su2[512];
  __shared__ float red[256];
  const int tid = threadIdx.x;

  for (int j = tid; j < cols; j += 256) su[j] = u[j];
  __syncthreads();
  for (int i = tid; i < rows; i += 256) {
    float acc = 0.f;
    const float* wr = W + (size_t)i * cols;
    for (int j = 0; j < cols; j++) acc += wr[j] * su[j];
    sv[i] = acc;
  }
  __syncthreads();
  float p = 0.f;
  for (int i = tid; i < rows; i += 256) p += sv[i] * sv[i];
  red[tid] = p; __syncthreads();
  for (int s = 128; s > 0; s >>= 1) { if (tid < s) red[tid] += red[tid + s]; __syncthreads(); }
  float inv = 1.0f / sqrtf(red[0]);
  __syncthreads();
  for (int i = tid; i < rows; i += 256) sv[i] *= inv;
  __syncthreads();
  for (int j = tid; j < cols; j += 256) {
    float acc = 0.f;
    for (int i = 0; i < rows; i++) acc += W[(size_t)i * cols + j] * sv[i];
    su2[j] = acc;
  }
  __syncthreads();
  p = 0.f;
  for (int j = tid; j < cols; j += 256) p += su2[j] * su2[j];
  red[tid] = p; __syncthreads();
  for (int s = 128; s > 0; s >>= 1) { if (tid < s) red[tid] += red[tid + s]; __syncthreads(); }
  float inv2 = 1.0f / sqrtf(red[0]);
  __syncthreads();
  for (int j = tid; j < cols; j += 256) su2[j] *= inv2;
  __syncthreads();
  p = 0.f;
  for (int i = tid; i < rows; i += 256) {
    float acc = 0.f;
    const float* wr = W + (size_t)i * cols;
    for (int j = 0; j < cols; j++) acc += wr[j] * su2[j];
    p += acc * sv[i];
  }
  red[tid] = p; __syncthreads();
  for (int s = 128; s > 0; s >>= 1) { if (tid < s) red[tid] += red[tid + s]; __syncthreads(); }
  if (tid == 0) out[0] = sigma[0] / red[0];
}

// ---------------------------------------------------------------------------
// Conversions to bf16 (scales folded into weights; exp(temperature) folded
// into the Q rows of W_qkv).
// ---------------------------------------------------------------------------
__global__ __launch_bounds__(256)
void cvt_x_kernel(const float* __restrict__ x, bf16* __restrict__ xb, int n) {
  int i = blockIdx.x * 256 + threadIdx.x;
  if (i < n) xb[i] = f2bf(x[i]);
}

__global__ __launch_bounds__(256)
void cvt_wqkv_kernel(const float* __restrict__ w, const float* __restrict__ scale,
                     const float* __restrict__ temperature, bf16* __restrict__ wb) {
  int i = blockIdx.x * 256 + threadIdx.x;     // 1536*512 total
  float s = scale[0];
  int row = i >> 9;                            // /512
  if (row < INNER) s *= __expf(temperature[0]);
  wb[i] = f2bf(w[i] * s);
}

__global__ __launch_bounds__(256)
void cvt_wout_kernel(const float* __restrict__ w, const float* __restrict__ scale,
                     bf16* __restrict__ wb, int n) {
  int i = blockIdx.x * 256 + threadIdx.x;
  if (i < n) wb[i] = f2bf(w[i] * scale[0]);
}

// ---------------------------------------------------------------------------
// K2: QKV projection. Y[8192,1536] = Xb @ Wb^T, written directly into
// Q/K/V [b,h,n,64] bf16. 8 waves/block; wave = 16 rows x 64 cols.
// ---------------------------------------------------------------------------
__global__ __launch_bounds__(256)
void qkv_gemm_kernel(const bf16* __restrict__ X, const bf16* __restrict__ W,
                     bf16* __restrict__ Q, bf16* __restrict__ K, bf16* __restrict__ V) {
  const int wid = threadIdx.x >> 5;
  const int lane = threadIdx.x & 31;
  const int g = lane >> 4, ln = lane & 15;
  const int wm = wid & 3, wn = wid >> 2;          // 4x2 waves
  const int row0 = blockIdx.x * 64 + wm * 16;
  const int col0 = blockIdx.y * 128 + wn * 64;

  v8f acc[4] = {};
  for (int k0 = 0; k0 < DIM; k0 += 32) {
    const bf16* ar = X + (size_t)(row0 + ln) * DIM + k0 + g * 16;
    v16bf a = make_a(*(const v8bf*)ar, *(const v8bf*)(ar + 8));
#pragma unroll
    for (int f = 0; f < 4; f++) {
      const bf16* br = W + (size_t)(col0 + f * 16 + ln) * DIM + k0 + g * 16;
      acc[f] = wmma_bf16(a, *(const v16bf*)br, acc[f]);
    }
  }
#pragma unroll
  for (int f = 0; f < 4; f++) {
#pragma unroll
    for (int r = 0; r < 8; r++) {
      const int R = row0 + r + 8 * g;
      const int C = col0 + f * 16 + ln;
      const int b = R >> 11, n = R & (N_ - 1);
      const int part = C >> 9, h = (C >> 6) & 7, d = C & 63;
      bf16* dst = (part == 0) ? Q : ((part == 1) ? K : V);
      dst[((size_t)(b * HEADS + h) * N_ + n) * DH + d] = f2bf(acc[f][r]);
    }
  }
}

// ---------------------------------------------------------------------------
// K3: V[bh,2048,64] -> Vt[bh,64,2048], tiled through LDS (coalesced both sides)
// ---------------------------------------------------------------------------
__global__ __launch_bounds__(256)
void transpose_v_kernel(const bf16* __restrict__ V, bf16* __restrict__ Vt) {
  __shared__ bf16 tile[64][65];
  const int bh = blockIdx.y;
  const int n0 = blockIdx.x * 64;
  const bf16* src = V + ((size_t)bh * N_ + n0) * DH;
  for (int idx = threadIdx.x; idx < 64 * 64; idx += 256) {
    int r = idx >> 6, c = idx & 63;
    tile[c][r] = src[(size_t)r * DH + c];
  }
  __syncthreads();
  bf16* dst = Vt + (size_t)bh * DH * N_ + n0;
  for (int idx = threadIdx.x; idx < 64 * 64; idx += 256) {
    int r = idx >> 6, c = idx & 63;
    dst[(size_t)r * N_ + c] = tile[r][c];
  }
}

// ---------------------------------------------------------------------------
// K4: flash attention, diagonal self-mask. Block = 4 waves, 64 query rows.
// Each wave owns 16 rows; online softmax with per-row (m,l) kept uniform
// across each 16-lane group via shfl_xor reductions. P re-laid-out C->A
// through a wave-private LDS slab (same-wave LDS is in-order).
// ---------------------------------------------------------------------------
__global__ __launch_bounds__(128)
void attn_kernel(const bf16* __restrict__ Q, const bf16* __restrict__ K,
                 const bf16* __restrict__ Vt, bf16* __restrict__ O) {
  __shared__ bf16 pbuf[4][16][32];
  const int wid = threadIdx.x >> 5;
  const int lane = threadIdx.x & 31;
  const int g = lane >> 4, ln = lane & 15;
  const int bh = blockIdx.y;
  const int row0 = blockIdx.x * 64 + wid * 16;
  const bf16* Qp = Q + (size_t)bh * N_ * DH;
  const bf16* Kp = K + (size_t)bh * N_ * DH;
  const bf16* Vp = Vt + (size_t)bh * DH * N_;

  v16bf qa[2];
#pragma unroll
  for (int kc = 0; kc < 2; kc++) {
    const bf16* p = Qp + (size_t)(row0 + ln) * DH + kc * 32 + g * 16;
    qa[kc] = make_a(*(const v8bf*)p, *(const v8bf*)(p + 8));
  }

  v8f o[4] = {};
  float mrow[8], lrow[8];
#pragma unroll
  for (int r = 0; r < 8; r++) { mrow[r] = NEG_MAX; lrow[r] = 0.f; }

  for (int j0 = 0; j0 < N_; j0 += 32) {
    v8f s[2] = {};
#pragma unroll
    for (int jc = 0; jc < 2; jc++) {
#pragma unroll
      for (int kc = 0; kc < 2; kc++) {
        const bf16* p = Kp + (size_t)(j0 + jc * 16 + ln) * DH + kc * 32 + g * 16;
        s[jc] = wmma_bf16(qa[kc], *(const v16bf*)p, s[jc]);
      }
    }
    // mask diagonal (self)
#pragma unroll
    for (int jc = 0; jc < 2; jc++) {
      const int col = j0 + jc * 16 + ln;
#pragma unroll
      for (int r = 0; r < 8; r++) {
        if ((row0 + r + 8 * g) == col) s[jc][r] = NEG_MAX;
      }
    }
    // online softmax: new row max
    float mnew[8], alpha[8];
#pragma unroll
    for (int r = 0; r < 8; r++) {
      float v = fmaxf(s[0][r], s[1][r]);
      v = fmaxf(v, __shfl_xor(v, 1));
      v = fmaxf(v, __shfl_xor(v, 2));
      v = fmaxf(v, __shfl_xor(v, 4));
      v = fmaxf(v, __shfl_xor(v, 8));
      mnew[r] = fmaxf(mrow[r], v);
      alpha[r] = __expf(mrow[r] - mnew[r]);
    }
    // p = exp(s - mnew); stage bf16 P into wave-private LDS (C->A relayout)
#pragma unroll
    for (int jc = 0; jc < 2; jc++) {
#pragma unroll
      for (int r = 0; r < 8; r++) {
        float pv = __expf(s[jc][r] - mnew[r]);
        s[jc][r] = pv;
        pbuf[wid][r + 8 * g][jc * 16 + ln] = f2bf(pv);
      }
    }
#pragma unroll
    for (int r = 0; r < 8; r++) {
      float ps = s[0][r] + s[1][r];
      ps += __shfl_xor(ps, 1);
      ps += __shfl_xor(ps, 2);
      ps += __shfl_xor(ps, 4);
      ps += __shfl_xor(ps, 8);
      lrow[r] = lrow[r] * alpha[r] + ps;
      mrow[r] = mnew[r];
    }
#pragma unroll
    for (int f = 0; f < 4; f++)
#pragma unroll
      for (int r = 0; r < 8; r++) o[f][r] *= alpha[r];

    // A-fragment of P from LDS, then O += P @ V
    const bf16* pb = &pbuf[wid][ln][g * 16];
    v16bf pa = make_a(*(const v8bf*)pb, *(const v8bf*)(pb + 8));
#pragma unroll
    for (int f = 0; f < 4; f++) {
      const bf16* vp = Vp + (size_t)(f * 16 + ln) * N_ + j0 + g * 16;
      o[f] = wmma_bf16(pa, *(const v16bf*)vp, o[f]);
    }
  }

  // finalize -> O[b, n, h*64] bf16
  const int b = bh >> 3, h = bh & 7;
#pragma unroll
  for (int r = 0; r < 8; r++) {
    const float invl = 1.0f / lrow[r];
    const int row = row0 + r + 8 * g;
#pragma unroll
    for (int f = 0; f < 4; f++) {
      O[((size_t)b * N_ + row) * INNER + h * DH + f * 16 + ln] = f2bf(o[f][r] * invl);
    }
  }
}

// ---------------------------------------------------------------------------
// K5: output projection. out[8192,512] = A @ Wo^T + bias   (fp32 out)
// ---------------------------------------------------------------------------
__global__ __launch_bounds__(256)
void out_gemm_kernel(const bf16* __restrict__ A, const bf16* __restrict__ W,
                     const float* __restrict__ bias, float* __restrict__ out) {
  const int wid = threadIdx.x >> 5;
  const int lane = threadIdx.x & 31;
  const int g = lane >> 4, ln = lane & 15;
  const int wm = wid & 3, wn = wid >> 2;
  const int row0 = blockIdx.x * 64 + wm * 16;
  const int col0 = blockIdx.y * 128 + wn * 64;

  v8f acc[4] = {};
  for (int k0 = 0; k0 < INNER; k0 += 32) {
    const bf16* ar = A + (size_t)(row0 + ln) * INNER + k0 + g * 16;
    v16bf a = make_a(*(const v8bf*)ar, *(const v8bf*)(ar + 8));
#pragma unroll
    for (int f = 0; f < 4; f++) {
      const bf16* br = W + (size_t)(col0 + f * 16 + ln) * INNER + k0 + g * 16;
      acc[f] = wmma_bf16(a, *(const v16bf*)br, acc[f]);
    }
  }
#pragma unroll
  for (int f = 0; f < 4; f++) {
#pragma unroll
    for (int r = 0; r < 8; r++) {
      const int R = row0 + r + 8 * g;
      const int C = col0 + f * 16 + ln;
      out[(size_t)R * DIM + C] = acc[f][r] + bias[C];
    }
  }
}

// ---------------------------------------------------------------------------
// Launcher. Inputs (setup_inputs order):
// 0 x[4,2048,512] f32   1 w_qkv[1536,512] f32   2 u_qkv[512]   3 sigma_qkv[1]
// 4 w_out[512,512]      5 b_out[512]            6 u_out[512]   7 sigma_out[1]
// 8 temperature[1]
// ---------------------------------------------------------------------------
extern "C" void kernel_launch(void* const* d_in, const int* in_sizes, int n_in,
                              void* d_out, int out_size, void* d_ws, size_t ws_size,
                              hipStream_t stream) {
  const float* x        = (const float*)d_in[0];
  const float* w_qkv    = (const float*)d_in[1];
  const float* u_qkv    = (const float*)d_in[2];
  const float* sig_qkv  = (const float*)d_in[3];
  const float* w_out    = (const float*)d_in[4];
  const float* b_out    = (const float*)d_in[5];
  const float* u_out    = (const float*)d_in[6];
  const float* sig_out  = (const float*)d_in[7];
  const float* temp     = (const float*)d_in[8];
  float* out = (float*)d_out;

  // workspace layout (all 256B aligned); total ~52.4 MB
  char* w = (char*)d_ws;
  float* scales = (float*)w;                       // [0]=qkv scale, [1]=out scale
  bf16* xb    = (bf16*)(w + 256);
  bf16* wqkvb = xb    + (size_t)ROWS * DIM;        // 8192*512
  bf16* woutb = wqkvb + (size_t)(3 * INNER) * DIM; // 1536*512
  bf16* Qb    = woutb + (size_t)DIM * INNER;       // 512*512
  bf16* Kb    = Qb    + (size_t)B_ * HEADS * N_ * DH;
  bf16* Vb    = Kb    + (size_t)B_ * HEADS * N_ * DH;
  bf16* Vtb   = Vb    + (size_t)B_ * HEADS * N_ * DH;
  bf16* attnb = Vtb   + (size_t)B_ * HEADS * N_ * DH;

  // 1) spectral-norm scales
  sn_scale_kernel<<<1, 256, 0, stream>>>(w_qkv, u_qkv, sig_qkv, scales + 0,
                                         3 * INNER, DIM);
  sn_scale_kernel<<<1, 256, 0, stream>>>(w_out, u_out, sig_out, scales + 1,
                                         DIM, INNER);
  // 2) bf16 conversions (scales folded; exp(temperature) folded into Q rows)
  {
    int n = ROWS * DIM;
    cvt_x_kernel<<<(n + 255) / 256, 256, 0, stream>>>(x, xb, n);
  }
  cvt_wqkv_kernel<<<(3 * INNER * DIM) / 256, 256, 0, stream>>>(w_qkv, scales + 0,
                                                               temp, wqkvb);
  {
    int n = DIM * INNER;
    cvt_wout_kernel<<<(n + 255) / 256, 256, 0, stream>>>(w_out, scales + 1, woutb, n);
  }
  // 3) QKV projection (WMMA) -> Q/K/V [b,h,n,64]
  qkv_gemm_kernel<<<dim3(ROWS / 64, (3 * INNER) / 128), 256, 0, stream>>>(
      xb, wqkvb, Qb, Kb, Vb);
  // 4) V transpose -> [b,h,64,n]
  transpose_v_kernel<<<dim3(N_ / 64, B_ * HEADS), 256, 0, stream>>>(Vb, Vtb);
  // 5) flash attention (WMMA, diag-masked, online softmax) -> [b,n,512] bf16
  attn_kernel<<<dim3(N_ / 64, B_ * HEADS), 128, 0, stream>>>(Qb, Kb, Vtb, attnb);
  // 6) output projection + bias (WMMA) -> fp32 d_out
  out_gemm_kernel<<<dim3(ROWS / 64, DIM / 128), 256, 0, stream>>>(
      attnb, woutb, b_out, out);

  (void)in_sizes; (void)n_in; (void)out_size; (void)ws_size;
}
